// DSQGAttention_9543417331806
// MI455X (gfx1250) — compile-verified, matrix operand order checked by reading, not run on hardware
//
#include <hip/hip_runtime.h>

typedef __attribute__((ext_vector_type(16))) __bf16 v16bf;
typedef __attribute__((ext_vector_type(8)))  __bf16 v8bf;
typedef __attribute__((ext_vector_type(4)))  __bf16 v4bf;
typedef __attribute__((ext_vector_type(8)))  float  v8f;

#define PITCH 40  // 32 + 8 pad bf16 elems: 80B row pitch (16B-aligned, conflict-free strides)

// C[M x N] = A[M x K] @ B[K x N], fp32 in/out, bf16 WMMA f32-accumulate.
// Grid: (M/128, N/128), block 256 = 8 waves in 2(M)x4(N); each wave: 64x32 strip
// = 4x2 tiles of 16x16 -> 8 WMMAs per 32-wide K-step. Ping-pong LDS, 1 barrier/step.
__global__ __launch_bounds__(256) void wmma_gemm_kernel(
    const float* __restrict__ A, const float* __restrict__ B,
    float* __restrict__ C, int N, int K) {
  __shared__ __attribute__((aligned(16))) __bf16 ldsA[2][128 * PITCH];
  __shared__ __attribute__((aligned(16))) __bf16 ldsBT[2][128 * PITCH];

  const int tid  = threadIdx.x;
  const int lane = tid & 31;
  const int wv   = tid >> 5;
  const int wm   = wv >> 2;   // 0..1 : 64-row M strip
  const int wn   = wv & 3;    // 0..3 : 32-col N strip
  const int mblock = blockIdx.x * 128;
  const int nblock = blockIdx.y * 128;

  const int half = lane >> 4;  // K-half select per ISA 16-bit fragment layout
  const int r16  = lane & 15;

  // A staging: thread stages row (tid>>1), 16 consecutive k at seg=(tid&1)*16
  const int s_arow = tid >> 1;
  const int s_aseg = (tid & 1) * 16;
  // B staging: thread stages cols [4*(tid&31),+4) x k-rows [4*(tid>>5),+4)
  const int s_bn = (tid & 31) * 4;
  const int s_bk = (tid >> 5) * 4;

  v8f acc[4][2];
#pragma unroll
  for (int i = 0; i < 4; ++i)
#pragma unroll
    for (int j = 0; j < 2; ++j) acc[i][j] = (v8f){};

  auto stage = [&](int buf, int k0) {
    // ---- A tile 128x32 -> bf16 LDS [row][k] ----
    const float* ap = A + (size_t)(mblock + s_arow) * K + (k0 + s_aseg);
    const float4 a0 = *reinterpret_cast<const float4*>(ap + 0);
    const float4 a1 = *reinterpret_cast<const float4*>(ap + 4);
    const float4 a2 = *reinterpret_cast<const float4*>(ap + 8);
    const float4 a3 = *reinterpret_cast<const float4*>(ap + 12);
    const v8bf wa0 = {(__bf16)a0.x, (__bf16)a0.y, (__bf16)a0.z, (__bf16)a0.w,
                      (__bf16)a1.x, (__bf16)a1.y, (__bf16)a1.z, (__bf16)a1.w};
    const v8bf wa1 = {(__bf16)a2.x, (__bf16)a2.y, (__bf16)a2.z, (__bf16)a2.w,
                      (__bf16)a3.x, (__bf16)a3.y, (__bf16)a3.z, (__bf16)a3.w};
    *reinterpret_cast<v8bf*>(&ldsA[buf][s_arow * PITCH + s_aseg + 0]) = wa0;
    *reinterpret_cast<v8bf*>(&ldsA[buf][s_arow * PITCH + s_aseg + 8]) = wa1;

    // ---- B tile 32x128 -> transposed bf16 LDS [n][k], packed b64 stores ----
    float g[4][4];
#pragma unroll
    for (int r = 0; r < 4; ++r) {
      const float4 v = *reinterpret_cast<const float4*>(
          B + (size_t)(k0 + s_bk + r) * N + (nblock + s_bn));
      g[r][0] = v.x; g[r][1] = v.y; g[r][2] = v.z; g[r][3] = v.w;
    }
#pragma unroll
    for (int i = 0; i < 4; ++i) {
      const v4bf col = {(__bf16)g[0][i], (__bf16)g[1][i],
                        (__bf16)g[2][i], (__bf16)g[3][i]};
      *reinterpret_cast<v4bf*>(&ldsBT[buf][(s_bn + i) * PITCH + s_bk]) = col;
    }
  };

  auto compute = [&](int buf) {
    // B fragments (2 per wave), reused across 4 A fragments
    v16bf bfrag[2];
#pragma unroll
    for (int fc = 0; fc < 2; ++fc) {
      const int nc = wn * 32 + fc * 16 + r16;
      const v8bf blo = *reinterpret_cast<const v8bf*>(&ldsBT[buf][nc * PITCH + half * 8]);
      const v8bf bhi = *reinterpret_cast<const v8bf*>(&ldsBT[buf][nc * PITCH + 16 + half * 8]);
#pragma unroll
      for (int i = 0; i < 8; ++i) { bfrag[fc][i] = blo[i]; bfrag[fc][i + 8] = bhi[i]; }
    }
#pragma unroll
    for (int fr = 0; fr < 4; ++fr) {
      const int mr = wm * 64 + fr * 16 + r16;
      const v8bf alo = *reinterpret_cast<const v8bf*>(&ldsA[buf][mr * PITCH + half * 8]);
      const v8bf ahi = *reinterpret_cast<const v8bf*>(&ldsA[buf][mr * PITCH + 16 + half * 8]);
      v16bf a;
#pragma unroll
      for (int i = 0; i < 8; ++i) { a[i] = alo[i]; a[i + 8] = ahi[i]; }
#pragma unroll
      for (int fc = 0; fc < 2; ++fc)
        acc[fr][fc] = __builtin_amdgcn_wmma_f32_16x16x32_bf16(
            false, a, false, bfrag[fc], (short)0, acc[fr][fc], false, false);
    }
  };

  stage(0, 0);
  int p = 0;
  for (int k0 = 0; k0 < K; k0 += 32) {
    __syncthreads();
    if (k0 + 32 < K) stage(p ^ 1, k0 + 32);
    compute(p);
    p ^= 1;
  }

  // ---- store: VGPR i -> row (i + 8*half), col r16 of each 16x16 tile ----
#pragma unroll
  for (int fr = 0; fr < 4; ++fr)
#pragma unroll
    for (int fc = 0; fc < 2; ++fc) {
      const int ncol = nblock + wn * 32 + fc * 16 + r16;
#pragma unroll
      for (int i = 0; i < 8; ++i) {
        const int m = mblock + wm * 64 + fr * 16 + i + 8 * half;
        C[(size_t)m * N + ncol] = acc[fr][fc][i];
      }
    }
}

__constant__ int c_offs[44] = {
    0, 1, 2, 3, 4, 5, 6, 7, 8, 9, 10, 11, 12, 13, 14, 15, 16, 17, 18, 19,
    20, 21, 22, 23, 24, 25, 26, 27, 28, 29, 30, 31, 32, 33, 34, 35, 36, 37,
    38, 39, 40, 128, 384, 1536};

// One 64-thread block per (t, h). qkv row t: [q(1024) | k(1024) | v(1024)].
__global__ __launch_bounds__(64) void attn_kernel(
    const float* __restrict__ qkv, const float* __restrict__ pos_bias,
    const float* __restrict__ scale_embed, const float* __restrict__ if_gain,
    float* __restrict__ out) {
  const int t = blockIdx.x;
  const int h = blockIdx.y;
  const int tid = threadIdx.x;  // 0..63

  __shared__ float qs[64];
  __shared__ float sc[64];

  qs[tid] = qkv[(size_t)t * 3072 + h * 64 + tid];
  __syncthreads();

  float s = -3.0e38f;
  if (tid < 44) {
    const int dj = c_offs[tid];
    if (t >= dj) {
      const float* krow = qkv + (size_t)(t - dj) * 3072 + 1024 + h * 64;
      const float* se = scale_embed + tid * 64;
      float acc = 0.f;
#pragma unroll 8
      for (int d = 0; d < 64; ++d) acc += qs[d] * krow[d] * (1.0f + se[d]);
      s = acc * 0.125f + pos_bias[tid * 16 + h];  // 0.125 = 1/sqrt(64)
    }
  }
  sc[tid] = s;
  __syncthreads();

  float m = -3.0e38f;
#pragma unroll 4
  for (int j = 0; j < 44; ++j) m = fmaxf(m, sc[j]);

  const int d = tid;
  float num = 0.f, den = 0.f;
  for (int j = 0; j < 44; ++j) {
    const int dj = c_offs[j];
    if (t >= dj) {
      const float w = __expf(sc[j] - m);
      den += w;
      num += w * qkv[(size_t)(t - dj) * 3072 + 2048 + h * 64 + d];
    }
  }
  out[(size_t)t * 1024 + h * 64 + d] = (num / den) * if_gain[h];
}

extern "C" void kernel_launch(void* const* d_in, const int* in_sizes, int n_in,
                              void* d_out, int out_size, void* d_ws, size_t ws_size,
                              hipStream_t stream) {
  (void)in_sizes; (void)n_in; (void)out_size; (void)ws_size;
  const float* x           = (const float*)d_in[0];  // (1, 2048, 1024)
  const float* w_qkv       = (const float*)d_in[1];  // (1024, 3072)
  const float* w_out       = (const float*)d_in[2];  // (1024, 1024)
  const float* pos_bias    = (const float*)d_in[3];  // (44, 16)
  const float* scale_embed = (const float*)d_in[4];  // (44, 64)
  const float* if_gain     = (const float*)d_in[5];  // (16,)
  float* out = (float*)d_out;                        // (1, 2048, 1024)

  const int T = 2048, C = 1024, H = 16;

  float* qkv     = (float*)d_ws;             // T * 3C floats = 25.2 MB
  float* attnout = qkv + (size_t)T * 3 * C;  // T * C floats  =  8.4 MB

  // qkv = x @ w_qkv : M=2048, N=3072, K=1024
  wmma_gemm_kernel<<<dim3(T / 128, (3 * C) / 128), 256, 0, stream>>>(
      x, w_qkv, qkv, 3 * C, C);

  // gathered-offset attention
  attn_kernel<<<dim3(T, H), 64, 0, stream>>>(qkv, pos_bias, scale_embed,
                                             if_gain, attnout);

  // out = attnout @ w_out : M=2048, N=1024, K=1024
  wmma_gemm_kernel<<<dim3(T / 128, C / 128), 256, 0, stream>>>(
      attnout, w_out, out, C, C);
}
